// WWS_SNN_Hardware_Sim_14405320310800
// MI455X (gfx1250) — 8-bit matrix operand order repaired and checked by reading, compiled, not run on hardware
//
#include <hip/hip_runtime.h>

typedef __attribute__((ext_vector_type(8))) int v8i;
typedef __attribute__((ext_vector_type(4))) int v4i;

#define BQ    4096
#define IN_D  1024
#define H1_D  1024
#define H2_D  512
#define OUT_D 35

#define BETA_F  (62259.0f / 65536.0f)
#define TH_F    0.5f
#define DEQ_F   (1.0f / 512.0f)

#define S_W1 (64 * 16 * 1024)
#define S_W2 (32 * 16 * 1024)
#define S_W3 (3 * 8 * 1024)

__device__ __forceinline__ int pack_lo(v4i w) {
  return (w.x & 0xFF) | ((w.y & 0xFF) << 8) | ((w.z & 0xFF) << 16) | ((w.w & 0xFF) << 24);
}
__device__ __forceinline__ int pack_hi(v4i w) {
  return ((w.x >> 8) & 0xFF) | (((w.y >> 8) & 0xFF) << 8) |
         (((w.z >> 8) & 0xFF) << 16) | (((w.w >> 8) & 0xFF) << 24);
}

__device__ __forceinline__ int a_koff(int v, int hv) {
  return ((v >> 2) << 5) + ((v & 3) << 2) + (hv << 4);
}

__global__ void pack_wplanes(const int* __restrict__ W, int Nrows, int Kdim,
                             int nTilesN, unsigned char* __restrict__ lo,
                             unsigned char* __restrict__ hi) {
  const int KT = Kdim >> 6;
  const int total = nTilesN * KT * 32;
  const int t = blockIdx.x * blockDim.x + threadIdx.x;
  if (t >= total) return;
  const int lane = t & 31;
  const int tile = t >> 5;
  const int kt = tile % KT;
  const int nt = tile / KT;
  const int hv = lane >> 4, l16 = lane & 15;
  const int o = nt * 16 + l16;
  const int valid = (o < Nrows);
  const int* wrow = W + (size_t)(valid ? o : 0) * Kdim + kt * 64 + hv * 16;
  v8i dlo = {0, 0, 0, 0, 0, 0, 0, 0};
  v8i dhi = {0, 0, 0, 0, 0, 0, 0, 0};
  if (valid) {
#pragma unroll
    for (int v = 0; v < 4; ++v) {
      const v4i w0 = *(const v4i*)(wrow + v * 4);
      const v4i w1 = *(const v4i*)(wrow + 32 + v * 4);
      dlo[v]     = pack_lo(w0);  dhi[v]     = pack_hi(w0);
      dlo[v + 4] = pack_lo(w1);  dhi[v + 4] = pack_hi(w1);
    }
  }
  const size_t off = (size_t)tile * 1024 + lane * 32;
  *(v8i*)(lo + off) = dlo;
  *(v8i*)(hi + off) = dhi;
}

struct AccPair { v8i lo, hi; };

__device__ __forceinline__ AccPair gemm_tile(const unsigned char* a_lds, int Kdim,
                                             const unsigned char* blo_t,
                                             const unsigned char* bhi_t,
                                             int lane, int hv, int l16) {
  v8i acc_lo = {0, 0, 0, 0, 0, 0, 0, 0};
  v8i acc_hi = {0, 0, 0, 0, 0, 0, 0, 0};
  const unsigned char* abase = a_lds + l16 * Kdim;
  const size_t lo32 = (size_t)lane * 32;
  for (int kt = 0; kt < (Kdim >> 6); ++kt) {
    const int k0 = kt << 6;
    __builtin_prefetch(blo_t + (size_t)(kt + 1) * 1024 + lo32, 0, 3);
    v8i a;
#pragma unroll
    for (int v = 0; v < 8; ++v)
      a[v] = *(const unsigned int*)(abase + k0 + a_koff(v, hv));
    const v8i blo = *(const v8i*)(blo_t + (size_t)kt * 1024 + lo32);
    const v8i bhi = *(const v8i*)(bhi_t + (size_t)kt * 1024 + lo32);
    acc_lo = __builtin_amdgcn_wmma_i32_16x16x64_iu8(false, a, false, blo, acc_lo, false, false);
    acc_hi = __builtin_amdgcn_wmma_i32_16x16x64_iu8(false, a, true,  bhi, acc_hi, false, false);
  }
  return {acc_lo, acc_hi};
}

__global__ __launch_bounds__(128)
void snn3_fused_iu8(const float* __restrict__ x,
                    const float* __restrict__ mem1_in,
                    const float* __restrict__ mem2_in,
                    const float* __restrict__ mem3_in,
                    const unsigned char* __restrict__ w1lo, const unsigned char* __restrict__ w1hi,
                    const unsigned char* __restrict__ w2lo, const unsigned char* __restrict__ w2hi,
                    const unsigned char* __restrict__ w3lo, const unsigned char* __restrict__ w3hi,
                    const int* __restrict__ B1, const int* __restrict__ B2,
                    const int* __restrict__ B3,
                    float* __restrict__ spk3_out,
                    float* __restrict__ mem1_out,
                    float* __restrict__ mem2_out,
                    float* __restrict__ mem3_out)
{
  __shared__ unsigned char s_xspk[16 * IN_D];
  __shared__ unsigned char s_spk1[16 * H1_D];
  __shared__ unsigned char s_spk2[16 * H2_D];

  const int lane = threadIdx.x & 31;
  const int wave = threadIdx.x >> 5;
  const int hv   = lane >> 4;
  const int l16  = lane & 15;
  const int m0   = blockIdx.x * 16;

  {
    const float* xblk = x + (size_t)m0 * IN_D;
    unsigned int* dst = (unsigned int*)s_xspk;
    for (int idx = threadIdx.x; idx < (16 * IN_D) / 4; idx += 128) {
      const float4 f = *(const float4*)(xblk + idx * 4);
      dst[idx] = ((int)f.x & 1) | (((int)f.y & 1) << 8) |
                 (((int)f.z & 1) << 16) | (((int)f.w & 1) << 24);
    }
  }
  __syncthreads();

  for (int nt = wave; nt < H1_D / 16; nt += 4) {
    const int ocol = nt * 16 + l16;
    const AccPair acc = gemm_tile(s_xspk, IN_D,
                                  w1lo + (size_t)nt * 16 * 1024,
                                  w1hi + (size_t)nt * 16 * 1024, lane, hv, l16);
    const int bias = B1[ocol];
#pragma unroll
    for (int r = 0; r < 8; ++r) {
      const int gm = m0 + r + (hv << 3);
      const float cur = (float)((acc.hi[r] << 8) + acc.lo[r] + bias) * DEQ_F;
      const float mo  = mem1_in[(size_t)gm * H1_D + ocol];
      const float mn  = BETA_F * mo + cur - ((mo > TH_F) ? TH_F : 0.0f);
      mem1_out[(size_t)gm * H1_D + ocol] = mn;
      s_spk1[(r + (hv << 3)) * H1_D + ocol] = (unsigned char)(mn > TH_F ? 1 : 0);
    }
  }
  __syncthreads();

  for (int nt = wave; nt < H2_D / 16; nt += 4) {
    const int ocol = nt * 16 + l16;
    const AccPair acc = gemm_tile(s_spk1, H1_D,
                                  w2lo + (size_t)nt * 16 * 1024,
                                  w2hi + (size_t)nt * 16 * 1024, lane, hv, l16);
    const int bias = B2[ocol];
#pragma unroll
    for (int r = 0; r < 8; ++r) {
      const int gm = m0 + r + (hv << 3);
      const float cur = (float)((acc.hi[r] << 8) + acc.lo[r] + bias) * DEQ_F;
      const float mo  = mem2_in[(size_t)gm * H2_D + ocol];
      const float mn  = BETA_F * mo + cur - ((mo > TH_F) ? TH_F : 0.0f);
      mem2_out[(size_t)gm * H2_D + ocol] = mn;
      s_spk2[(r + (hv << 3)) * H2_D + ocol] = (unsigned char)(mn > TH_F ? 1 : 0);
    }
  }
  __syncthreads();

  for (int nt = wave; nt < 3; nt += 4) {
    const int ocol = nt * 16 + l16;
    const AccPair acc = gemm_tile(s_spk2, H2_D,
                                  w3lo + (size_t)nt * 8 * 1024,
                                  w3hi + (size_t)nt * 8 * 1024, lane, hv, l16);
    if (ocol < OUT_D) {
      const int bias = B3[ocol];
#pragma unroll
      for (int r = 0; r < 8; ++r) {
        const int gm = m0 + r + (hv << 3);
        const float cur = (float)((acc.hi[r] << 8) + acc.lo[r] + bias) * DEQ_F;
        const float mo  = mem3_in[(size_t)gm * OUT_D + ocol];
        const float mn  = BETA_F * mo + cur - ((mo > TH_F) ? TH_F : 0.0f);
        mem3_out[(size_t)gm * OUT_D + ocol] = mn;
        spk3_out[(size_t)gm * OUT_D + ocol] = (mn > TH_F) ? 1.0f : 0.0f;
      }
    }
  }
}

extern "C" void kernel_launch(void* const* d_in, const int* in_sizes, int n_in,
                              void* d_out, int out_size, void* d_ws, size_t ws_size,
                              hipStream_t stream) {
  (void)in_sizes; (void)n_in; (void)out_size; (void)ws_size;
  const float* x     = (const float*)d_in[0];
  const float* mem1  = (const float*)d_in[1];
  const float* mem2  = (const float*)d_in[2];
  const float* mem3  = (const float*)d_in[3];
  const int*   W1    = (const int*)d_in[4];
  const int*   B1    = (const int*)d_in[5];
  const int*   W2    = (const int*)d_in[6];
  const int*   B2    = (const int*)d_in[7];
  const int*   W3    = (const int*)d_in[8];
  const int*   B3    = (const int*)d_in[9];

  unsigned char* ws = (unsigned char*)d_ws;
  unsigned char* w1lo = ws;
  unsigned char* w1hi = w1lo + S_W1;
  unsigned char* w2lo = w1hi + S_W1;
  unsigned char* w2hi = w2lo + S_W2;
  unsigned char* w3lo = w2hi + S_W2;
  unsigned char* w3hi = w3lo + S_W3;

  float* out    = (float*)d_out;
  float* spk3_o = out;
  float* mem1_o = spk3_o + (size_t)BQ * OUT_D;
  float* mem2_o = mem1_o + (size_t)BQ * H1_D;
  float* mem3_o = mem2_o + (size_t)BQ * H2_D;

  pack_wplanes<<<dim3((64 * 16 * 32) / 256), dim3(256), 0, stream>>>(W1, H1_D, IN_D, 64, w1lo, w1hi);
  pack_wplanes<<<dim3((32 * 16 * 32) / 256), dim3(256), 0, stream>>>(W2, H2_D, H1_D, 32, w2lo, w2hi);
  pack_wplanes<<<dim3(3, 1, 1), dim3(256), 0, stream>>>(W3, OUT_D, H2_D, 3, w3lo, w3hi);

  snn3_fused_iu8<<<dim3(BQ / 16), dim3(128), 0, stream>>>(
      x, mem1, mem2, mem3, w1lo, w1hi, w2lo, w2hi, w3lo, w3hi,
      B1, B2, B3, spk3_o, mem1_o, mem2_o, mem3_o);
}
